// Tmp_attention3_70841190580534
// MI455X (gfx1250) — matrix operand order repaired and checked by reading, not compiled, not run on hardware
//
#include <hip/hip_runtime.h>
#include <hip/hip_bf16.h>
#include <math.h>

// ---------------- problem constants ----------------
#define BB 2
#define NN 1024
#define CS 384
#define HH 16
#define CC 64
#define PVV 8
#define VC 96            // 64 v-channels + 24 rotated point channels + 8 pad
#define CATD 1536        // H*C + H*PV*4
#define INF_C 100000.0f
#define EPS_C 1e-8f
// 2 * sqrt(1/(3*C))  (logits = a_qk + mask + a_qk)
#define TWOSCALE 0.14433756729740643f

typedef __attribute__((ext_vector_type(16))) __bf16 bf16x16;
typedef __attribute__((ext_vector_type(8)))  __bf16 bf16x8;
typedef __attribute__((ext_vector_type(8)))  float  v8f;

// ---------------- WMMA helper ----------------
__device__ __forceinline__ v8f wmma_bf16(bf16x16 a, bf16x16 b, v8f c) {
  return __builtin_amdgcn_wmma_f32_16x16x32_bf16(false, a, false, b, (short)0, c,
                                                 false, false);
}

// A-fragment (16x32, bf16), per ISA layout: lane m = lane&15, half h = lane>>4.
// frag[e] = row[kbase + 8h + e] (e<8), frag[8+e] = row[kbase + 16 + 8h + e]
__device__ __forceinline__ bf16x16 ld_afrag(const __bf16* row, int kbase, int hf) {
  const __bf16* p = row + kbase + 8 * hf;
  bf16x8 lo = *(const bf16x8*)p;
  bf16x8 hi = *(const bf16x8*)(p + 16);
  bf16x16 r;
#pragma unroll
  for (int e = 0; e < 8; ++e) { r[e] = lo[e]; r[e + 8] = hi[e]; }
  return r;
}

// B-fragment (32x16, bf16): lane n = lane&15, half h. frag[e] = col[8h + e] (e<8),
// frag[8+e] = col[16 + 8h + e], where col points at element (k=kbase, n) of a layout with K contiguous.
__device__ __forceinline__ bf16x16 ld_bfrag(const __bf16* col, int hf) {
  const bf16x8* q = (const bf16x8*)(col + 8 * hf);
  bf16x8 lo = q[0], hi = q[2];
  bf16x16 r;
#pragma unroll
  for (int e = 0; e < 8; ++e) { r[e] = lo[e]; r[e + 8] = hi[e]; }
  return r;
}

__device__ __forceinline__ float xor16f(float v) {
  int lane = threadIdx.x & 31;
  return __int_as_float(
      __builtin_amdgcn_ds_bpermute(((lane ^ 16) << 2), __float_as_int(v)));
}
__device__ __forceinline__ float bcastf(float v, int src) {
  return __int_as_float(__builtin_amdgcn_ds_bpermute(src << 2, __float_as_int(v)));
}

// ---------------- kernel 0a: transpose-convert weight f32(K x N) -> bf16(N x K) --
__global__ __launch_bounds__(256) void wcvt_kernel(const float* __restrict__ W,
                                                   __bf16* __restrict__ Wt,
                                                   int Kdim, int Ncols) {
  const int idx = blockIdx.x * 256 + threadIdx.x;       // n*Kdim + k
  const int n = idx / Kdim, k = idx - n * Kdim;
  Wt[idx] = (__bf16)W[(size_t)k * Ncols + n];
}
// ---------------- kernel 0b: straight convert f32 -> bf16 -------------------
__global__ __launch_bounds__(256) void scvt_kernel(const float* __restrict__ S,
                                                   __bf16* __restrict__ Sbf) {
  const int idx = blockIdx.x * 256 + threadIdx.x;
  Sbf[idx] = (__bf16)S[idx];
}

// ---------------- kernel 1: projection GEMM (s @ W + b), bf16 WMMA ----------------
// Each wave computes a 16x64 output tile: one A-frag feeds 4 WMMAs per k-step.
__global__ __launch_bounds__(256) void proj_gemm(
    const __bf16* __restrict__ Sbf, const __bf16* __restrict__ Wt,
    const float* __restrict__ bias, int Ncols, int mode,
    __bf16* __restrict__ Qbf, __bf16* __restrict__ Kbf,
    __bf16* __restrict__ Vt, float* __restrict__ VPraw) {
  const int wave = threadIdx.x >> 5, lane = threadIdx.x & 31;
  const int il = lane & 15, hf = lane >> 4;
  const int tile = blockIdx.x * 8 + wave;
  const int ng = Ncols >> 6;                 // groups of 64 columns
  const int tm = tile / ng, tg = tile % ng;
  const int m0 = tm << 4, n0 = tg << 6;

  const v8f vz = {0.f, 0.f, 0.f, 0.f, 0.f, 0.f, 0.f, 0.f};
  v8f acc[4];
#pragma unroll
  for (int s = 0; s < 4; ++s) acc[s] = vz;

  const __bf16* Arow = Sbf + (size_t)(m0 + il) * CS;
  for (int k0 = 0; k0 < CS; k0 += 32) {
    const bf16x16 af = ld_afrag(Arow, k0, hf);
#pragma unroll
    for (int s = 0; s < 4; ++s) {
      const __bf16* bcol = Wt + (size_t)(n0 + s * 16 + il) * CS + k0;
      acc[s] = wmma_bf16(af, ld_bfrag(bcol, hf), acc[s]);
    }
  }

#pragma unroll
  for (int s = 0; s < 4; ++s) {
    const int col = n0 + s * 16 + il;
    const float bv = bias[col];
#pragma unroll
    for (int r = 0; r < 8; ++r) {
      const int m = m0 + 8 * hf + r;
      const int b = m >> 10, n = m & (NN - 1);
      const float y = acc[s][r] + bv;
      if (mode == 0) {                       // q -> Q[b][h][n][c] bf16
        int h = col >> 6, c = col & 63;
        Qbf[(((size_t)(b * HH + h) * NN) + n) * CC + c] = (__bf16)y;
      } else if (mode == 1) {                // kv -> K[b][h][n][c] / Vt[b][h][ch][n]
        int h = col >> 7, rr = col & 127;
        if (rr < CC)
          Kbf[(((size_t)(b * HH + h) * NN) + n) * CC + rr] = (__bf16)y;
        else
          Vt[(((size_t)(b * HH + h) * VC) + (rr - CC)) * NN + n] = (__bf16)y;
      } else {                               // vp -> raw f32 for rotation
        VPraw[((size_t)b * NN + n) * 384 + col] = y;
      }
    }
  }
}

// ---------------- kernel 2: vp rotation + mask bias + pad ----------------
__global__ __launch_bounds__(256) void prep_kernel(
    const float* __restrict__ VPraw, const float* __restrict__ rot,
    const float* __restrict__ trans, const float* __restrict__ mask,
    float* __restrict__ maskbias, __bf16* __restrict__ Vt) {
  const int idx = blockIdx.x * 256 + threadIdx.x;   // B*N*256
  const int p2 = idx & 255, bn = idx >> 8;
  const int b = bn >> 10, n = bn & (NN - 1);
  if (p2 == 0) maskbias[bn] = INF_C * (mask[bn] - 1.f);
  if (p2 < 128) {
    const int h = p2 >> 3, pv = p2 & 7;
    const float v0 = VPraw[(size_t)bn * 384 + p2];
    const float v1 = VPraw[(size_t)bn * 384 + 128 + p2];
    const float v2 = VPraw[(size_t)bn * 384 + 256 + p2];
    const float* Rm = rot + (size_t)bn * 9;
    const float* Tv = trans + (size_t)bn * 3;
#pragma unroll
    for (int i = 0; i < 3; ++i) {
      float o = Rm[i * 3 + 0] * v0 + Rm[i * 3 + 1] * v1 + Rm[i * 3 + 2] * v2 + Tv[i];
      Vt[(((size_t)(b * HH + h) * VC) + (64 + i * 8 + pv)) * NN + n] = (__bf16)o;
    }
  } else {
    const int q = p2 - 128, h = q >> 3, z = q & 7;
    Vt[(((size_t)(b * HH + h) * VC) + (88 + z)) * NN + n] = (__bf16)0.f;
  }
}

// ---------------- kernel 3: flash attention (streaming softmax, WMMA) ----------------
// One wave handles 16 query rows; computes S^T tiles (A=K, B=Q) so the P re-pack
// is lane-local; accumulates O (96 ch = v ++ rotated-vp) over all j.
__global__ __launch_bounds__(256) void attn_kernel(
    const __bf16* __restrict__ Q, const __bf16* __restrict__ K,
    const __bf16* __restrict__ Vt, const float* __restrict__ maskbias,
    float* __restrict__ Oatt) {
  const int wave = threadIdx.x >> 5, lane = threadIdx.x & 31;
  const int il = lane & 15, hf = lane >> 4;
  const int bh = blockIdx.x >> 3;         // N/128 = 8 tiles per (b,h)
  const int tile = blockIdx.x & 7;
  const int b = bh / HH;
  const int i0 = tile * 128 + wave * 16;

  const __bf16* Qb = Q + (size_t)bh * NN * CC;
  const __bf16* Kb = K + (size_t)bh * NN * CC;
  const __bf16* Vb = Vt + (size_t)bh * VC * NN;
  float* Ob = Oatt + (size_t)bh * NN * VC;
  const float* mb = maskbias + b * NN;

  const int ig = i0 + il;
  const __bf16* qrow = Qb + (size_t)ig * CC;
  const bf16x16 qf0 = ld_bfrag(qrow, hf);        // c 0..31
  const bf16x16 qf1 = ld_bfrag(qrow + 32, hf);   // c 32..63
  const float biasi = mb[ig];

  float mrun = -3.0e38f, lrun = 0.f;
  v8f oacc[6];
  const v8f vz = {0.f, 0.f, 0.f, 0.f, 0.f, 0.f, 0.f, 0.f};
#pragma unroll
  for (int t = 0; t < 6; ++t) oacc[t] = vz;

  for (int j0 = 0; j0 < NN; j0 += 64) {
    if (j0 + 64 < NN) {                          // gfx1250 global_prefetch_b8
      const __bf16* pk = Kb + (size_t)(j0 + 64) * CC;
      __builtin_prefetch(pk + lane * 128, 0, 0);
      __builtin_prefetch(pk + lane * 128 + 64, 0, 0);
    }
    float pt[4][8];
    float mloc = -3.0e38f;
#pragma unroll
    for (int sub = 0; sub < 4; ++sub) {
      const __bf16* krow = Kb + (size_t)(j0 + sub * 16 + il) * CC;
      v8f s = vz;
      s = wmma_bf16(ld_afrag(krow, 0, hf), qf0, s);
      s = wmma_bf16(ld_afrag(krow, 32, hf), qf1, s);
      const float* mj = mb + j0 + sub * 16 + 8 * hf;   // bias per j
#pragma unroll
      for (int r = 0; r < 8; ++r) {
        float lg = TWOSCALE * s[r] + biasi + mj[r];
        pt[sub][r] = lg;
        mloc = fmaxf(mloc, lg);
      }
    }
    mloc = fmaxf(mloc, xor16f(mloc));            // combine the two j-halves
    const float mnew = fmaxf(mrun, mloc);
    const float alpha = __expf(mrun - mnew);
    float psum = 0.f;
#pragma unroll
    for (int sub = 0; sub < 4; ++sub)
#pragma unroll
      for (int r = 0; r < 8; ++r) {
        float p = __expf(pt[sub][r] - mnew);
        pt[sub][r] = p;
        psum += p;
      }
    psum += xor16f(psum);
    lrun = lrun * alpha + psum;
    mrun = mnew;

    // lane-local repack of P into A-fragments (j 0..31 and 32..63)
    bf16x16 pa0, pa1;
#pragma unroll
    for (int e = 0; e < 8; ++e) {
      pa0[e] = (__bf16)pt[0][e];  pa0[e + 8] = (__bf16)pt[1][e];
      pa1[e] = (__bf16)pt[2][e];  pa1[e + 8] = (__bf16)pt[3][e];
    }
    float af[8];
#pragma unroll
    for (int r = 0; r < 8; ++r) af[r] = bcastf(alpha, 8 * hf + r);
#pragma unroll
    for (int t = 0; t < 6; ++t) {
#pragma unroll
      for (int r = 0; r < 8; ++r) oacc[t][r] *= af[r];
      const __bf16* vc = Vb + (size_t)(t * 16 + il) * NN + j0;
      oacc[t] = wmma_bf16(pa0, ld_bfrag(vc, hf), oacc[t]);
      oacc[t] = wmma_bf16(pa1, ld_bfrag(vc + 32, hf), oacc[t]);
    }
  }

  float lf[8];
  const float linv = 1.f / lrun;
#pragma unroll
  for (int r = 0; r < 8; ++r) lf[r] = bcastf(linv, 8 * hf + r);
#pragma unroll
  for (int t = 0; t < 6; ++t)
#pragma unroll
    for (int r = 0; r < 8; ++r)
      Ob[(size_t)(i0 + 8 * hf + r) * VC + t * 16 + il] = oacc[t][r] * lf[r];
}

// ---------------- kernel 4: epilogue transform -> cat (bf16) ----------------
__global__ __launch_bounds__(256) void epi1_kernel(
    const float* __restrict__ Oatt, const float* __restrict__ rot,
    const float* __restrict__ trans, __bf16* __restrict__ catbf) {
  const int idx = blockIdx.x * 256 + threadIdx.x;  // B*N*H*PV
  const int p = idx & 7, t = idx >> 3;
  const int h = t & 15, bn = t >> 4;
  const float* Orow = Oatt + (((size_t)((bn >> 10) * HH + h) * NN) + (bn & (NN - 1))) * VC;
  __bf16* crow = catbf + (size_t)bn * CATD;
#pragma unroll
  for (int c = 0; c < 8; ++c)                       // copy o channels
    crow[h * CC + p * 8 + c] = (__bf16)Orow[p * 8 + c];
  const float* Tv = trans + (size_t)bn * 3;
  const float d0 = Orow[64 + 0 * 8 + p] - Tv[0];
  const float d1 = Orow[64 + 1 * 8 + p] - Tv[1];
  const float d2 = Orow[64 + 2 * 8 + p] - Tv[2];
  const float* Rm = rot + (size_t)bn * 9;           // out_i = sum_j rot[j][i]*d_j
  const float w0 = Rm[0] * d0 + Rm[3] * d1 + Rm[6] * d2;
  const float w1 = Rm[1] * d0 + Rm[4] * d1 + Rm[7] * d2;
  const float w2 = Rm[2] * d0 + Rm[5] * d1 + Rm[8] * d2;
  const float nm = sqrtf(w0 * w0 + w1 * w1 + w2 * w2 + EPS_C);
  const int hp = h * PVV + p;
  crow[1024 + hp]       = (__bf16)w0;
  crow[1024 + 128 + hp] = (__bf16)w1;
  crow[1024 + 256 + hp] = (__bf16)w2;
  crow[1024 + 384 + hp] = (__bf16)nm;
}

// ---------------- kernel 5: output GEMM cat @ w_out + b_out ----------------
// 16x64 per wave; Wt_out is bf16 (N x K) so B-frags are contiguous.
__global__ __launch_bounds__(256) void out_gemm(
    const __bf16* __restrict__ catbf, const __bf16* __restrict__ Wt,
    const float* __restrict__ bout, float* __restrict__ out) {
  const int wave = threadIdx.x >> 5, lane = threadIdx.x & 31;
  const int il = lane & 15, hf = lane >> 4;
  const int tile = blockIdx.x * 8 + wave;
  const int tm = tile / 6, tg = tile % 6;           // 128 x 6 tiles of 16x64
  const int m0 = tm << 4, n0 = tg << 6;

  const v8f vz = {0.f, 0.f, 0.f, 0.f, 0.f, 0.f, 0.f, 0.f};
  v8f acc[4];
#pragma unroll
  for (int s = 0; s < 4; ++s) acc[s] = vz;

  const __bf16* Arow = catbf + (size_t)(m0 + il) * CATD;
  for (int k0 = 0; k0 < CATD; k0 += 32) {
    const bf16x16 af = ld_afrag(Arow, k0, hf);
#pragma unroll
    for (int s = 0; s < 4; ++s) {
      const __bf16* bcol = Wt + (size_t)(n0 + s * 16 + il) * CATD + k0;
      acc[s] = wmma_bf16(af, ld_bfrag(bcol, hf), acc[s]);
    }
  }
#pragma unroll
  for (int s = 0; s < 4; ++s) {
    const int col = n0 + s * 16 + il;
    const float bv = bout[col];
#pragma unroll
    for (int r = 0; r < 8; ++r)
      out[(size_t)(m0 + 8 * hf + r) * CS + col] = acc[s][r] + bv;
  }
}

// ---------------- launcher ----------------
extern "C" void kernel_launch(void* const* d_in, const int* in_sizes, int n_in,
                              void* d_out, int out_size, void* d_ws, size_t ws_size,
                              hipStream_t stream) {
  const float* s     = (const float*)d_in[0];
  // d_in[1] = z : UNUSED by the reference (64 MB we never touch)
  const float* rot   = (const float*)d_in[2];
  const float* trans = (const float*)d_in[3];
  const float* mask  = (const float*)d_in[4];
  const float* w_q   = (const float*)d_in[5];
  const float* b_q   = (const float*)d_in[6];
  const float* w_kv  = (const float*)d_in[7];
  const float* b_kv  = (const float*)d_in[8];
  const float* w_vp  = (const float*)d_in[9];
  const float* b_vp  = (const float*)d_in[10];
  const float* w_out = (const float*)d_in[11];
  const float* b_out = (const float*)d_in[12];
  float* out = (float*)d_out;

  char* ws = (char*)d_ws;
  size_t off = 0;
  __bf16* Qbf = (__bf16*)(ws + off); off += (size_t)BB * HH * NN * CC * 2;   // 4 MiB
  __bf16* Kbf = (__bf16*)(ws + off); off += (size_t)BB * HH * NN * CC * 2;   // 4 MiB
  __bf16* Vt  = (__bf16*)(ws + off); off += (size_t)BB * HH * VC * NN * 2;   // 6 MiB
  float*  VPr = (float*)(ws + off);  off += (size_t)BB * NN * 384 * 4;       // 3 MiB
  float*  mbs = (float*)(ws + off);  off += (size_t)BB * NN * 4;             // 8 KiB
  float*  Oat = (float*)(ws + off);  off += (size_t)BB * HH * NN * VC * 4;   // 12 MiB
  __bf16* cat = (__bf16*)(ws + off); off += (size_t)BB * NN * CATD * 2;      // 6 MiB
  __bf16* Sbf = (__bf16*)(ws + off); off += (size_t)BB * NN * CS * 2;        // 1.5 MiB
  __bf16* WqT = (__bf16*)(ws + off); off += (size_t)1024 * CS * 2;           // 0.75 MiB
  __bf16* WkT = (__bf16*)(ws + off); off += (size_t)2048 * CS * 2;           // 1.5 MiB
  __bf16* WvT = (__bf16*)(ws + off); off += (size_t)384 * CS * 2;            // 0.28 MiB
  __bf16* WoT = (__bf16*)(ws + off); off += (size_t)CS * CATD * 2;           // 1.1 MiB
  (void)ws_size; (void)in_sizes; (void)n_in; (void)out_size;

  dim3 blk(256);
  // weight/activation bf16 conversion (weights transposed to N x K)
  scvt_kernel<<<(BB * NN * CS) / 256, blk, 0, stream>>>(s, Sbf);
  wcvt_kernel<<<(1024 * CS) / 256, blk, 0, stream>>>(w_q, WqT, CS, 1024);
  wcvt_kernel<<<(2048 * CS) / 256, blk, 0, stream>>>(w_kv, WkT, CS, 2048);
  wcvt_kernel<<<(384 * CS) / 256, blk, 0, stream>>>(w_vp, WvT, CS, 384);
  wcvt_kernel<<<(CS * CATD) / 256, blk, 0, stream>>>(w_out, WoT, CATD, CS);
  // projections: q (1024 cols), kv (2048 cols), vp (384 cols); 16x64 per wave
  proj_gemm<<<(128 * 16) / 8, blk, 0, stream>>>(Sbf, WqT, b_q, 1024, 0, Qbf, Kbf, Vt, VPr);
  proj_gemm<<<(128 * 32) / 8, blk, 0, stream>>>(Sbf, WkT, b_kv, 2048, 1, Qbf, Kbf, Vt, VPr);
  proj_gemm<<<(128 * 6) / 8, blk, 0, stream>>>(Sbf, WvT, b_vp, 384, 2, Qbf, Kbf, Vt, VPr);
  // vp rotation + mask bias + V pad
  prep_kernel<<<(BB * NN * 256) / 256, blk, 0, stream>>>(VPr, rot, trans, mask, mbs, Vt);
  // flash attention over fused 96-channel V'
  attn_kernel<<<BB * HH * (NN / 128), blk, 0, stream>>>(Qbf, Kbf, Vt, mbs, Oat);
  // inverse-frame transform + norms + concat
  epi1_kernel<<<(BB * NN * HH * PVV) / 256, blk, 0, stream>>>(Oat, rot, trans, cat);
  // final projection
  out_gemm<<<(128 * 6) / 8, blk, 0, stream>>>(cat, WoT, b_out, out);
}
